// AbsolutePE_LM_41661182771539
// MI455X (gfx1250) — compile-verified, matrix operand order checked by reading, not run on hardware
//
#include <hip/hip_runtime.h>
#include <hip/hip_bf16.h>

#define B_SZ  4
#define CTX   2048
#define DM    1024
#define VOCAB 32000

typedef __attribute__((ext_vector_type(16))) _Float16 v16h;
typedef __attribute__((ext_vector_type(8)))  float    v8f;

#define LSTR 40   // padded LDS row stride in halves (32 data + 8 pad = 80B rows)

// ---------------------------------------------------------------------------
// Async-copy one 128x32 f16 tile (row-major, leading dim ld) from global into
// an LDS tile with padded stride LSTR. 512 16B chunks over 256 threads (two
// branchless issues per thread -> no EXEC masking), each issued as
// GLOBAL_LOAD_ASYNC_TO_LDS_B128 (ASYNCcnt-tracked, GV addressing: VDST holds
// the LDS byte address, VADDR the 64-bit global address).
// ---------------------------------------------------------------------------
__device__ __forceinline__ void async_chunk(const _Float16* __restrict__ g,
                                            int ld, int row0, int k0,
                                            _Float16* lds, int c) {
    const int row = c >> 2;
    const int w   = (c & 3) * 8;                           // half offset in row
    const _Float16* gp = g + (size_t)(row0 + row) * (size_t)ld + k0 + w;
    const unsigned  lp = (unsigned)(uintptr_t)(lds + row * LSTR + w);
    asm volatile("global_load_async_to_lds_b128 %0, %1, off"
                 :: "v"(lp), "v"(gp) : "memory");
}

__device__ __forceinline__ void async_tile(const _Float16* __restrict__ g,
                                           int ld, int row0, int k0,
                                           _Float16* lds, int tid) {
    async_chunk(g, ld, row0, k0, lds, tid);
    async_chunk(g, ld, row0, k0, lds, tid + 256);
}

// ---------------------------------------------------------------------------
// Read one 16x32 f16 WMMA fragment from an LDS tile (padded stride LSTR).
// CDNA5 16-bit A/B layout: lane half h holds K = h*8+{0..7} (VGPR0-3) and
// K = 16+h*8+{0..7} (VGPR4-7) for row = lane&15 -> two ds_load_b128 per frag.
// ---------------------------------------------------------------------------
__device__ __forceinline__ v16h lds_frag(const _Float16* tile, int row, int lane) {
    const int r = lane & 15;
    const int h = lane >> 4;
    const _Float16* p = tile + (row + r) * LSTR + h * 8;
    v16h f;
    *reinterpret_cast<uint4*>(&f)       = *reinterpret_cast<const uint4*>(p);
    *(reinterpret_cast<uint4*>(&f) + 1) = *reinterpret_cast<const uint4*>(p + 16);
    return f;
}

// One 32-wide k-step: 4 A frags + 2 B frags from LDS, 8 WMMAs.
__device__ __forceinline__ void wmma_step(const _Float16* As, const _Float16* Bs,
                                          v8f acc[4][2], int wrow, int wcol, int lane) {
    v16h a[4], b[2];
#pragma unroll
    for (int i = 0; i < 4; ++i) a[i] = lds_frag(As, wrow * 64 + 16 * i, lane);
#pragma unroll
    for (int j = 0; j < 2; ++j) b[j] = lds_frag(Bs, wcol * 32 + 16 * j, lane);
#pragma unroll
    for (int i = 0; i < 4; ++i)
#pragma unroll
        for (int j = 0; j < 2; ++j)
            acc[i][j] = __builtin_amdgcn_wmma_f32_16x16x32_f16(
                false, a[i], false, b[j], (short)0, acc[i][j], false, false);
}

// ---------------------------------------------------------------------------
// Generic WMMA GEMM: C[M,N] = A[M,K] * Bt[N,K]^T   (A, Bt row-major f16)
// 256 threads = 8 waves; block tile 128x128; wave tile 64x32 (4x2 WMMA tiles).
// Double-buffered async LDS pipeline, k-loop unrolled x2 so all LDS buffer
// offsets are static immediates. Requires K % 64 == 0 (holds: K = 1024/2048).
// MODE 0: f16 out, row-major                      (Q, K projections)
// MODE 1: f16 out, per-batch transposed [B][D][C] (V -> Vt)
// MODE 2: f32 out * alpha, batched via blockIdx.z, causal block skip (scores)
// MODE 3: f16 out = f16(acc + Add), batched       (attn_out + residual)
// MODE 4: f32 out = acc + Bias[n]                 (vocab logits)
// ---------------------------------------------------------------------------
template <int MODE>
__global__ __launch_bounds__(256)
void gemm_wmma(const _Float16* __restrict__ A, const _Float16* __restrict__ Bt,
               void* __restrict__ Out, const float* __restrict__ Add,
               const float* __restrict__ Bias,
               int M, int N, int K, int lda, int ldb, int ldc, float alpha)
{
    if (MODE == 2 && blockIdx.x > blockIdx.y) return;   // fully-masked causal block

    __shared__ _Float16 As[2][128 * LSTR];
    __shared__ _Float16 Bs[2][128 * LSTR];

    const int tid  = threadIdx.x;
    const int lane = tid & 31;
    const int wave = tid >> 5;
    const int wrow = wave >> 2;   // 0..1
    const int wcol = wave & 3;    // 0..3

    const int mblock = blockIdx.y * 128;
    const int nblock = blockIdx.x * 128;

    size_t zA = 0, zB = 0;
    if (MODE == 2) { zA = (size_t)blockIdx.z * CTX * DM;  zB = (size_t)blockIdx.z * CTX * DM; }
    if (MODE == 3) { zA = (size_t)blockIdx.z * CTX * CTX; zB = (size_t)blockIdx.z * DM * CTX; }
    const _Float16* __restrict__ Ab = A + zA;
    const _Float16* __restrict__ Bb = Bt + zB;

    // Prologue: stream first k-tile into buffer 0.
    async_tile(Ab, lda, mblock, 0, &As[0][0], tid);
    async_tile(Bb, ldb, nblock, 0, &Bs[0][0], tid);

    v8f acc[4][2] = {};

    for (int k0 = 0; k0 < K; k0 += 64) {
        // --- stage 0: compute buf0, prefetch k0+32 into buf1 ---
        asm volatile("s_wait_asynccnt 0x0" ::: "memory");
        __syncthreads();
        if (k0 + 32 < K) {
            async_tile(Ab, lda, mblock, k0 + 32, &As[1][0], tid);
            async_tile(Bb, ldb, nblock, k0 + 32, &Bs[1][0], tid);
        }
        wmma_step(&As[0][0], &Bs[0][0], acc, wrow, wcol, lane);

        // --- stage 1: compute buf1, prefetch k0+64 into buf0 ---
        asm volatile("s_wait_asynccnt 0x0" ::: "memory");
        __syncthreads();
        if (k0 + 64 < K) {
            async_tile(Ab, lda, mblock, k0 + 64, &As[0][0], tid);
            async_tile(Bb, ldb, nblock, k0 + 64, &Bs[0][0], tid);
        }
        wmma_step(&As[1][0], &Bs[1][0], acc, wrow, wcol, lane);
    }

    // Epilogue. C/D layout: lane half hh, col n = n0+16j+(lane&15),
    // row m = m0+16i + v + 8*hh for accumulator element v.
    const int m0 = mblock + wrow * 64;
    const int n0 = nblock + wcol * 32;
    const int r  = lane & 15;
    const int hh = lane >> 4;
#pragma unroll
    for (int i = 0; i < 4; ++i) {
#pragma unroll
        for (int j = 0; j < 2; ++j) {
            const int mb = m0 + 16 * i + 8 * hh;
            const int n  = n0 + 16 * j + r;
#pragma unroll
            for (int v = 0; v < 8; ++v) {
                const int m = mb + v;
                float val = acc[i][j][v];
                if (MODE == 0) {
                    ((_Float16*)Out)[(size_t)m * ldc + n] = (_Float16)val;
                } else if (MODE == 1) {
                    const int bb = m >> 11;            // m / CTX
                    const int c  = m & (CTX - 1);      // m % CTX
                    ((_Float16*)Out)[(((size_t)bb * DM) + n) * CTX + c] = (_Float16)val;
                } else if (MODE == 2) {
                    ((float*)Out)[(size_t)blockIdx.z * CTX * CTX + (size_t)m * ldc + n] = val * alpha;
                } else if (MODE == 3) {
                    const size_t idx = ((size_t)blockIdx.z * CTX + m) * (size_t)DM + n;
                    ((_Float16*)Out)[idx] = (_Float16)(val + Add[idx]);
                } else { // MODE == 4
                    ((float*)Out)[(size_t)m * ldc + n] = val + Bias[n];
                }
            }
        }
    }
}

// ---------------------------------------------------------------------------
// f32 [R][C] -> f16 [C][R] transpose-convert (LDS tiled, conflict-free)
// ---------------------------------------------------------------------------
__global__ __launch_bounds__(256)
void transpose_f32_to_f16(const float* __restrict__ src, _Float16* __restrict__ dst,
                          int R, int Cc) {
    __shared__ float tile[32][33];
    const int c0 = blockIdx.x * 32;
    const int r0 = blockIdx.y * 32;
    const int tx = threadIdx.x;   // 0..31
    const int ty = threadIdx.y;   // 0..7
#pragma unroll
    for (int i = 0; i < 32; i += 8)
        tile[ty + i][tx] = src[(size_t)(r0 + ty + i) * Cc + c0 + tx];
    __syncthreads();
#pragma unroll
    for (int i = 0; i < 32; i += 8)
        dst[(size_t)(c0 + ty + i) * R + r0 + tx] = (_Float16)tile[tx][ty + i];
}

// ---------------------------------------------------------------------------
// h = embed_W[x] + pos_W   -> f32 (residual) and f16 (WMMA operand)
// ---------------------------------------------------------------------------
__global__ __launch_bounds__(256)
void embed_kernel(const int* __restrict__ x, const float* __restrict__ embW,
                  const float* __restrict__ posW,
                  float* __restrict__ h32, _Float16* __restrict__ h16) {
    const int bc  = blockIdx.x;            // 0 .. B*CTX-1
    const int c   = bc & (CTX - 1);
    const int tok = x[bc];
    const float* e  = embW + (size_t)tok * DM;
    const float* pp = posW + (size_t)c * DM;
    const size_t base = (size_t)bc * DM;
    for (int d = threadIdx.x; d < DM; d += blockDim.x) {
        const float v = e[d] + pp[d];
        h32[base + d] = v;
        h16[base + d] = (_Float16)v;
    }
}

// ---------------------------------------------------------------------------
// Row-wise causal softmax over f32 scores -> f16 probabilities (zeros above diag)
// ---------------------------------------------------------------------------
__global__ __launch_bounds__(256)
void softmax_kernel(const float* __restrict__ scores, _Float16* __restrict__ P) {
    const int q = blockIdx.x;
    const int b = blockIdx.y;
    const float* row  = scores + ((size_t)b * CTX + q) * (size_t)CTX;
    _Float16*    prow = P      + ((size_t)b * CTX + q) * (size_t)CTX;
    __shared__ float red[256];
    const int tid = threadIdx.x;

    float m = -3.4e38f;
    for (int k = tid; k <= q; k += 256) m = fmaxf(m, row[k]);
    red[tid] = m; __syncthreads();
    for (int s = 128; s > 0; s >>= 1) { if (tid < s) red[tid] = fmaxf(red[tid], red[tid + s]); __syncthreads(); }
    m = red[0]; __syncthreads();

    float sum = 0.0f;
    for (int k = tid; k <= q; k += 256) sum += __expf(row[k] - m);
    red[tid] = sum; __syncthreads();
    for (int s = 128; s > 0; s >>= 1) { if (tid < s) red[tid] += red[tid + s]; __syncthreads(); }
    const float inv = 1.0f / red[0];

    for (int k = tid; k < CTX; k += 256)
        prow[k] = (k <= q) ? (_Float16)(__expf(row[k] - m) * inv) : (_Float16)0.0f;
}

// ---------------------------------------------------------------------------
extern "C" void kernel_launch(void* const* d_in, const int* in_sizes, int n_in,
                              void* d_out, int out_size, void* d_ws, size_t ws_size,
                              hipStream_t stream) {
    (void)in_sizes; (void)n_in; (void)out_size; (void)ws_size;
    const int*   x    = (const int*)  d_in[0];
    const float* embW = (const float*)d_in[1];
    const float* posW = (const float*)d_in[2];
    const float* Wq   = (const float*)d_in[3];
    const float* Wk   = (const float*)d_in[4];
    const float* Wv   = (const float*)d_in[5];
    const float* outW = (const float*)d_in[6];
    const float* outb = (const float*)d_in[7];
    float* out = (float*)d_out;

    char* w = (char*)d_ws;
    auto take = [&](size_t bytes) -> char* {
        char* p = w; w += (bytes + 255) & ~(size_t)255; return p;
    };
    float*     h32 = (float*)    take((size_t)B_SZ * CTX * DM * 4);
    _Float16*  h16 = (_Float16*) take((size_t)B_SZ * CTX * DM * 2);
    _Float16*  WqT = (_Float16*) take((size_t)DM * DM * 2);
    _Float16*  WkT = (_Float16*) take((size_t)DM * DM * 2);
    _Float16*  WvT = (_Float16*) take((size_t)DM * DM * 2);
    _Float16*  oWT = (_Float16*) take((size_t)DM * VOCAB * 2);
    _Float16*  Q   = (_Float16*) take((size_t)B_SZ * CTX * DM * 2);
    _Float16*  Kb  = (_Float16*) take((size_t)B_SZ * CTX * DM * 2);
    _Float16*  Vt  = (_Float16*) take((size_t)B_SZ * DM * CTX * 2);
    float*     S   = (float*)    take((size_t)B_SZ * CTX * CTX * 4);
    _Float16*  P   = (_Float16*) take((size_t)B_SZ * CTX * CTX * 2);
    _Float16*  hpa = (_Float16*) take((size_t)B_SZ * CTX * DM * 2);

    const dim3 tb(32, 8);
    // Weight transpose-converts (B operands stored [N][K] so WMMA B frags are contiguous)
    transpose_f32_to_f16<<<dim3(DM / 32,    DM / 32), tb, 0, stream>>>(Wq,   WqT, DM, DM);
    transpose_f32_to_f16<<<dim3(DM / 32,    DM / 32), tb, 0, stream>>>(Wk,   WkT, DM, DM);
    transpose_f32_to_f16<<<dim3(DM / 32,    DM / 32), tb, 0, stream>>>(Wv,   WvT, DM, DM);
    transpose_f32_to_f16<<<dim3(VOCAB / 32, DM / 32), tb, 0, stream>>>(outW, oWT, DM, VOCAB);

    // h = embed + pos
    embed_kernel<<<B_SZ * CTX, 256, 0, stream>>>(x, embW, posW, h32, h16);

    const int M = B_SZ * CTX;   // 8192
    // Q, K projections (f16 out)
    gemm_wmma<0><<<dim3(DM / 128, M / 128), 256, 0, stream>>>(h16, WqT, Q,  nullptr, nullptr, M, DM, DM, DM, DM, DM, 1.0f);
    gemm_wmma<0><<<dim3(DM / 128, M / 128), 256, 0, stream>>>(h16, WkT, Kb, nullptr, nullptr, M, DM, DM, DM, DM, DM, 1.0f);
    // V projection written transposed per batch: Vt[b][d][c]
    gemm_wmma<1><<<dim3(DM / 128, M / 128), 256, 0, stream>>>(h16, WvT, Vt, nullptr, nullptr, M, DM, DM, DM, DM, 0,  1.0f);
    // scores[b] = (Q K^T) / sqrt(D), causal blocks skipped
    gemm_wmma<2><<<dim3(CTX / 128, CTX / 128, B_SZ), 256, 0, stream>>>(Q, Kb, S, nullptr, nullptr, CTX, CTX, DM, DM, DM, CTX, 0.03125f);
    // causal softmax -> P (f16)
    softmax_kernel<<<dim3(CTX, B_SZ), 256, 0, stream>>>(S, P);
    // attn_out = P @ V, fused residual: hpa = f16(h + attn_out)
    gemm_wmma<3><<<dim3(DM / 128, CTX / 128, B_SZ), 256, 0, stream>>>(P, Vt, hpa, h32, nullptr, CTX, DM, CTX, CTX, CTX, DM, 1.0f);
    // logits = hpa @ out_W + out_b  (the 550-GFLOP GEMM; out_W^T f16 is L2-resident)
    gemm_wmma<4><<<dim3(VOCAB / 128, M / 128), 256, 0, stream>>>(hpa, oWT, out, nullptr, outb, M, VOCAB, DM, DM, DM, VOCAB, 1.0f);
}
